// SelfAttention_84258668413059
// MI455X (gfx1250) — matrix operand order repaired and checked by reading, not compiled, not run on hardware
//
#include <hip/hip_runtime.h>

typedef __bf16 bf16;
typedef __attribute__((ext_vector_type(8)))  __bf16 v8bf;
typedef __attribute__((ext_vector_type(16))) __bf16 v16bf;
typedef __attribute__((ext_vector_type(8)))  float  v8f;

#define B_  4
#define D_  1024
#define N_  2048
#define MH_ 1024
#define MV_ 1024

// ---------------- fp32 -> bf16 flat convert ----------------
__global__ void f2bf_kernel(const float* __restrict__ in, bf16* __restrict__ out, int n) {
  for (int i = blockIdx.x * blockDim.x + threadIdx.x; i < n; i += gridDim.x * blockDim.x)
    out[i] = (bf16)in[i];
}

// ------- transpose+convert x[b][D][N] f32 -> xT[b][N][D] bf16 -------
__global__ __launch_bounds__(256) void xpose_kernel(const float* __restrict__ x,
                                                    bf16* __restrict__ xT) {
  __shared__ float tile[32][33];
  const int b  = blockIdx.z;
  const int d0 = blockIdx.y * 32;
  const int n0 = blockIdx.x * 32;
  const float* xb  = x  + (size_t)b * D_ * N_;
  bf16*        xTb = xT + (size_t)b * N_ * D_;
  const int tx = threadIdx.x;   // 0..31
  const int ty = threadIdx.y;   // 0..7
  #pragma unroll
  for (int i = 0; i < 32; i += 8)
    tile[ty + i][tx] = xb[(size_t)(d0 + ty + i) * N_ + (n0 + tx)];
  __syncthreads();
  #pragma unroll
  for (int i = 0; i < 32; i += 8)
    xTb[(size_t)(n0 + ty + i) * D_ + (d0 + tx)] = (bf16)tile[tx][ty + i];
}

// ---------------- NT GEMM: C[m,n] = sum_k A[m,k] * B[n,k] ----------------
// A: [M,K] row-major bf16 (lda = K stride), B: [N,K] row-major bf16 (ldb).
// OUT_MODE 0: bf16 transposed store  Ct[n*ldc + m]   (packed b128 stores)
// OUT_MODE 1: bf16 natural store     C[m*ldc + n]    (elementwise)
// OUT_MODE 2: f32  natural store     C[m*ldc + n]    (elementwise)
// Block = 128x128 C-tile, 256 threads = 8 waves (wave32), 4(M) x 2(N) wave grid.
// Each wave owns 32(M) x 64(N) = 2x4 WMMA 16x16 tiles. K-step = 32.
// Data movement: GLOBAL_LOAD_ASYNC_TO_LDS_B128 ping-pong double buffer,
// ASYNCcnt-synchronized, one workgroup barrier per K-step.
template<int OUT_MODE>
__global__ __launch_bounds__(256) void gemm_nt(const bf16* __restrict__ A,
                                               const bf16* __restrict__ Bm,
                                               void* __restrict__ Cv,
                                               int K, int lda, int ldb, int ldc,
                                               size_t sA, size_t sB, size_t sC) {
  constexpr int KT = 32;
  constexpr int KP = 40;                 // padded K pitch (elements) -> 80B rows
  __shared__ bf16 As[2][128 * KP];
  __shared__ bf16 Bs[2][128 * KP];

  const int t    = threadIdx.x;
  const int lane = t & 31;
  const int wave = t >> 5;
  const int wm   = wave >> 1;            // 0..3
  const int wn   = wave & 1;             // 0..1
  const int l16  = lane & 15;
  const int hl   = lane >> 4;            // 0/1

  const int m0 = blockIdx.y * 128;
  const int n0 = blockIdx.x * 128;
  const int b  = blockIdx.z;

  const bf16* Ab = A  + (size_t)b * sA;
  const bf16* Bb = Bm + (size_t)b * sB;
  const unsigned long long AbU = (unsigned long long)(size_t)Ab;
  const unsigned long long BbU = (unsigned long long)(size_t)Bb;

  // Per-thread copy slots: 512 x 16B chunks per 128x32 tile, 2 per thread.
  const int c0 = t,        rowA0 = c0 >> 2, ko0 = (c0 & 3) * 8;
  const int c1 = t + 256,  rowA1 = c1 >> 2, ko1 = (c1 & 3) * 8;
  // Global byte offsets excluding the k0 term (added per step).
  const unsigned gA0 = (unsigned)(((m0 + rowA0) * lda + ko0) * 2);
  const unsigned gA1 = (unsigned)(((m0 + rowA1) * lda + ko1) * 2);
  const unsigned gB0 = (unsigned)(((n0 + rowA0) * ldb + ko0) * 2);
  const unsigned gB1 = (unsigned)(((n0 + rowA1) * ldb + ko1) * 2);

  auto issue_tile = [&](int buf, int k0) {
    const unsigned kB = (unsigned)(k0 * 2);
    const unsigned lA0 = (unsigned)(size_t)&As[buf][rowA0 * KP + ko0];
    const unsigned lA1 = (unsigned)(size_t)&As[buf][rowA1 * KP + ko1];
    const unsigned lB0 = (unsigned)(size_t)&Bs[buf][rowA0 * KP + ko0];
    const unsigned lB1 = (unsigned)(size_t)&Bs[buf][rowA1 * KP + ko1];
    const unsigned oA0 = gA0 + kB, oA1 = gA1 + kB, oB0 = gB0 + kB, oB1 = gB1 + kB;
    asm volatile("global_load_async_to_lds_b128 %0, %1, %2"
                 :: "v"(lA0), "v"(oA0), "s"(AbU) : "memory");
    asm volatile("global_load_async_to_lds_b128 %0, %1, %2"
                 :: "v"(lA1), "v"(oA1), "s"(AbU) : "memory");
    asm volatile("global_load_async_to_lds_b128 %0, %1, %2"
                 :: "v"(lB0), "v"(oB0), "s"(BbU) : "memory");
    asm volatile("global_load_async_to_lds_b128 %0, %1, %2"
                 :: "v"(lB1), "v"(oB1), "s"(BbU) : "memory");
  };

  v8f acc[2][4];
  #pragma unroll
  for (int mi = 0; mi < 2; ++mi)
    #pragma unroll
    for (int nj = 0; nj < 4; ++nj)
      acc[mi][nj] = (v8f){0.f, 0.f, 0.f, 0.f, 0.f, 0.f, 0.f, 0.f};

  // Prologue: prefetch first K-slice into buffer 0.
  issue_tile(0, 0);

  const int nsteps = K / KT;
  for (int it = 0; it < nsteps; ++it) {
    const int p = it & 1;
    // My slice of buffer p has landed in LDS; barrier makes everyone's visible
    // and guarantees all waves finished reading the other buffer.
    asm volatile("s_wait_asynccnt 0x0" ::: "memory");
    __syncthreads();
    if (it + 1 < nsteps) issue_tile(p ^ 1, (it + 1) * KT);

    // Fragments from buffer p. A 16x32 per-lane K striping: lanes<16 hold
    // K {0..7, 16..23}, lanes>=16 hold K {8..15, 24..31}. B 32x16: same K
    // striping as A (column = lane&15).
    v16bf af[2], bfr[4];
    #pragma unroll
    for (int mi = 0; mi < 2; ++mi) {
      const int r = wm * 32 + mi * 16 + l16;
      v8bf lo = *(const v8bf*)&As[p][r * KP + 8 * hl];
      v8bf hi = *(const v8bf*)&As[p][r * KP + 8 * hl + 16];
      af[mi] = __builtin_shufflevector(lo, hi, 0,1,2,3,4,5,6,7,8,9,10,11,12,13,14,15);
    }
    #pragma unroll
    for (int nj = 0; nj < 4; ++nj) {
      const int r = wn * 64 + nj * 16 + l16;
      v8bf lo = *(const v8bf*)&Bs[p][r * KP + 8 * hl];
      v8bf hi = *(const v8bf*)&Bs[p][r * KP + 8 * hl + 16];
      bfr[nj] = __builtin_shufflevector(lo, hi, 0,1,2,3,4,5,6,7,8,9,10,11,12,13,14,15);
    }

    #pragma unroll
    for (int mi = 0; mi < 2; ++mi)
      #pragma unroll
      for (int nj = 0; nj < 4; ++nj)
        acc[mi][nj] = __builtin_amdgcn_wmma_f32_16x16x32_bf16(
            false, af[mi], false, bfr[nj], (short)0, acc[mi][nj], false, false);
  }

  // D layout: lane holds column n = (lane&15); VGPR r -> row m = r + 8*(lane>>4).
  if constexpr (OUT_MODE == 0) {
    bf16* Ct = (bf16*)Cv + (size_t)b * sC;
    #pragma unroll
    for (int mi = 0; mi < 2; ++mi)
      #pragma unroll
      for (int nj = 0; nj < 4; ++nj) {
        const int n_g = n0 + wn * 64 + nj * 16 + l16;
        const int m_b = m0 + wm * 32 + mi * 16 + 8 * hl;
        v8bf pk;
        #pragma unroll
        for (int r = 0; r < 8; ++r) pk[r] = (bf16)acc[mi][nj][r];
        *(v8bf*)&Ct[(size_t)n_g * ldc + m_b] = pk;     // 16B packed store along m
      }
  } else if constexpr (OUT_MODE == 1) {
    bf16* Cn = (bf16*)Cv + (size_t)b * sC;
    #pragma unroll
    for (int mi = 0; mi < 2; ++mi)
      #pragma unroll
      for (int nj = 0; nj < 4; ++nj) {
        const int n_g = n0 + wn * 64 + nj * 16 + l16;
        #pragma unroll
        for (int r = 0; r < 8; ++r) {
          const int m_g = m0 + wm * 32 + mi * 16 + 8 * hl + r;
          Cn[(size_t)m_g * ldc + n_g] = (bf16)acc[mi][nj][r];
        }
      }
  } else {
    float* Cf = (float*)Cv + (size_t)b * sC;
    #pragma unroll
    for (int mi = 0; mi < 2; ++mi)
      #pragma unroll
      for (int nj = 0; nj < 4; ++nj) {
        const int n_g = n0 + wn * 64 + nj * 16 + l16;
        #pragma unroll
        for (int r = 0; r < 8; ++r) {
          const int m_g = m0 + wm * 32 + mi * 16 + 8 * hl + r;
          Cf[(size_t)m_g * ldc + n_g] = acc[mi][nj][r];
        }
      }
  }
}

// ------- softmax over j for fixed i, on ST[b][j][i] (in place, bf16) -------
__global__ __launch_bounds__(256) void softmax_cols_kernel(bf16* __restrict__ S) {
  const int b  = blockIdx.y;
  const int i0 = blockIdx.x * 64;
  bf16* Sb = S + (size_t)b * N_ * N_;
  const int t  = threadIdx.x;
  const int ii = t & 63;
  const int jg = t >> 6;                 // 0..3
  const int i  = i0 + ii;

  float m = -3.0e38f, s = 0.f;
  for (int j = jg; j < N_; j += 4) {
    const float v = (float)Sb[(size_t)j * N_ + i];
    const float nm = fmaxf(m, v);
    s = s * __expf(m - nm) + __expf(v - nm);
    m = nm;
  }
  __shared__ float rm[4][64], rs[4][64];
  rm[jg][ii] = m; rs[jg][ii] = s;
  __syncthreads();
  float M = -3.0e38f, T = 0.f;
  #pragma unroll
  for (int g = 0; g < 4; ++g) {
    const float gm = rm[g][ii], gs = rs[g][ii];
    const float nm = fmaxf(M, gm);
    T = T * __expf(M - nm) + gs * __expf(gm - nm);
    M = nm;
  }
  const float inv = 1.0f / T;
  for (int j = jg; j < N_; j += 4) {
    const size_t idx = (size_t)j * N_ + i;
    const float v = (float)Sb[idx];
    Sb[idx] = (bf16)(__expf(v - M) * inv);
  }
}

extern "C" void kernel_launch(void* const* d_in, const int* in_sizes, int n_in,
                              void* d_out, int out_size, void* d_ws, size_t ws_size,
                              hipStream_t stream) {
  const float* x  = (const float*)d_in[0];
  const float* WQ = (const float*)d_in[1];
  const float* WK = (const float*)d_in[2];
  const float* WV = (const float*)d_in[3];
  const float* WO = (const float*)d_in[4];
  float* out = (float*)d_out;

  // Workspace layout (bf16 everywhere, ~120 MB total)
  char* w = (char*)d_ws;
  bf16* xT = (bf16*)w; w += (size_t)B_ * N_ * D_ * 2;    // x^T  [b][n][d]
  bf16* wq = (bf16*)w; w += (size_t)MH_ * D_ * 2;
  bf16* wk = (bf16*)w; w += (size_t)MH_ * D_ * 2;
  bf16* wv = (bf16*)w; w += (size_t)MV_ * D_ * 2;
  bf16* wo = (bf16*)w; w += (size_t)D_ * MV_ * 2;
  bf16* qT = (bf16*)w; w += (size_t)B_ * N_ * MH_ * 2;   // q^T  [b][n][h]
  bf16* kT = (bf16*)w; w += (size_t)B_ * N_ * MH_ * 2;   // k^T  [b][n][h]
  bf16* vp = (bf16*)w; w += (size_t)B_ * MV_ * N_ * 2;   // v    [b][v][n]
  bf16* St = (bf16*)w; w += (size_t)B_ * N_ * N_ * 2;    // S^T  [b][j][i] -> P^T
  bf16* aT = (bf16*)w; w += (size_t)B_ * N_ * MV_ * 2;   // attn^T [b][n][v]
  if ((size_t)(w - (char*)d_ws) > ws_size) return;       // ws too small: bail

  // 1) precision conversion
  f2bf_kernel<<<1024, 256, 0, stream>>>(WQ, wq, MH_ * D_);
  f2bf_kernel<<<1024, 256, 0, stream>>>(WK, wk, MH_ * D_);
  f2bf_kernel<<<1024, 256, 0, stream>>>(WV, wv, MV_ * D_);
  f2bf_kernel<<<1024, 256, 0, stream>>>(WO, wo, D_ * MV_);
  xpose_kernel<<<dim3(N_ / 32, D_ / 32, B_), dim3(32, 8), 0, stream>>>(x, xT);

  // 2) projections: C[h,n] = sum_d W[h,d] * xT[n,d]
  gemm_nt<0><<<dim3(N_ / 128, MH_ / 128, B_), 256, 0, stream>>>(
      wq, xT, qT, D_, D_, D_, MH_, 0, (size_t)N_ * D_, (size_t)N_ * MH_);
  gemm_nt<0><<<dim3(N_ / 128, MH_ / 128, B_), 256, 0, stream>>>(
      wk, xT, kT, D_, D_, D_, MH_, 0, (size_t)N_ * D_, (size_t)N_ * MH_);
  gemm_nt<1><<<dim3(N_ / 128, MV_ / 128, B_), 256, 0, stream>>>(
      wv, xT, vp, D_, D_, D_, N_, 0, (size_t)N_ * D_, (size_t)MV_ * N_);

  // 3) scores: S[i,j] = sum_h kT[i,h] * qT[j,h]; stored transposed -> ST[j][i]
  gemm_nt<0><<<dim3(N_ / 128, N_ / 128, B_), 256, 0, stream>>>(
      kT, qT, St, MH_, MH_, MH_, N_,
      (size_t)N_ * MH_, (size_t)N_ * MH_, (size_t)N_ * N_);

  // 4) softmax over j (rows of ST), in place -> P^T[j][i]
  softmax_cols_kernel<<<dim3(N_ / 64, B_), 256, 0, stream>>>(St);

  // 5) attn[v,j] = sum_i v[v,i] * PT[j,i]; stored transposed -> attnT[j][v]
  gemm_nt<0><<<dim3(N_ / 128, MV_ / 128, B_), 256, 0, stream>>>(
      vp, St, aT, N_, N_, N_, MV_,
      (size_t)MV_ * N_, (size_t)N_ * N_, (size_t)N_ * MV_);

  // 6) out[d,n] = sum_v WO[d,v] * attnT[n,v]; natural f32 store to d_out
  gemm_nt<2><<<dim3(N_ / 128, D_ / 128, B_), 256, 0, stream>>>(
      wo, aT, out, MV_, MV_, MV_, N_,
      0, (size_t)N_ * MV_, (size_t)D_ * N_);
}
